// Down_4054449128183
// MI455X (gfx1250) — matrix operand order repaired and checked by reading, not compiled, not run on hardware
//
#include <hip/hip_runtime.h>
#include <hip/hip_bf16.h>
#include <math.h>

// Problem constants (from reference)
#define B_    4
#define N_    8192
#define CIN_  64
#define COUT_ 64
#define M_    2048      // N / STRIDE
#define MKER_ 8
#define KNN_  16
#define SNH_  16

typedef __attribute__((ext_vector_type(16))) _Float16 v16h;
typedef __attribute__((ext_vector_type(8)))  _Float16 v8h;
typedef __attribute__((ext_vector_type(8)))  float    v8f;

// ---------------------------------------------------------------------------
// Kernel 1: furthest point sampling. One 1024-thread block per batch.
// Each thread owns 8 points (coords + running min-dist in registers).
// Per step: register argmax -> wave32 shfl reduce -> cross-wave LDS reduce.
// Also writes the p2 output (gathered xyz) directly.
// ---------------------------------------------------------------------------
__global__ __launch_bounds__(1024) void fps_kernel(const float* __restrict__ xyz,
                                                   int* __restrict__ idx,
                                                   float* __restrict__ p2)
{
    const int b = blockIdx.x;
    const int t = threadIdx.x;
    const int lane = t & 31, wave = t >> 5;
    const float* X = xyz + (size_t)b * N_ * 3;

    float px[8], py[8], pz[8], md[8];
    const float x0 = X[0], y0 = X[1], z0 = X[2];
#pragma unroll
    for (int s = 0; s < 8; ++s) {
        const int p = t + 1024 * s;
        px[s] = X[p * 3 + 0]; py[s] = X[p * 3 + 1]; pz[s] = X[p * 3 + 2];
        const float dx = px[s] - x0, dy = py[s] - y0, dz = pz[s] - z0;
        md[s] = dx * dx + dy * dy + dz * dz;
    }

    __shared__ float wv[32];
    __shared__ int   wi[32];
    __shared__ float bc[3];
    __shared__ int   bidx;

    if (t == 0) {
        idx[b * M_] = 0;
        p2[((size_t)b * M_) * 3 + 0] = x0;
        p2[((size_t)b * M_) * 3 + 1] = y0;
        p2[((size_t)b * M_) * 3 + 2] = z0;
    }

    for (int it = 1; it < M_; ++it) {
        // per-thread argmax over owned points
        float best = -1.0f; int bp = 0;
#pragma unroll
        for (int s = 0; s < 8; ++s)
            if (md[s] > best) { best = md[s]; bp = t + 1024 * s; }
        // wave32 reduce (lowest index wins ties, matching jnp.argmax)
#pragma unroll
        for (int off = 16; off > 0; off >>= 1) {
            const float ov = __shfl_xor(best, off, 32);
            const int   oi = __shfl_xor(bp, off, 32);
            if (ov > best || (ov == best && oi < bp)) { best = ov; bp = oi; }
        }
        if (lane == 0) { wv[wave] = best; wi[wave] = bp; }
        __syncthreads();
        if (wave == 0) {
            float v = wv[lane]; int i2 = wi[lane];
#pragma unroll
            for (int off = 16; off > 0; off >>= 1) {
                const float ov = __shfl_xor(v, off, 32);
                const int   oi = __shfl_xor(i2, off, 32);
                if (ov > v || (ov == v && oi < i2)) { v = ov; i2 = oi; }
            }
            if (lane == 0) bidx = i2;
        }
        __syncthreads();
        const int p = bidx;
        if (t == (p & 1023)) {            // owner thread broadcasts coords
            const int s = p >> 10;
            bc[0] = px[s]; bc[1] = py[s]; bc[2] = pz[s];
            idx[b * M_ + it] = p;
            p2[((size_t)b * M_ + it) * 3 + 0] = px[s];
            p2[((size_t)b * M_ + it) * 3 + 1] = py[s];
            p2[((size_t)b * M_ + it) * 3 + 2] = pz[s];
        }
        __syncthreads();
        const float qx = bc[0], qy = bc[1], qz = bc[2];
#pragma unroll
        for (int s = 0; s < 8; ++s) {
            const float dx = px[s] - qx, dy = py[s] - qy, dz = pz[s] - qz;
            md[s] = fminf(md[s], dx * dx + dy * dy + dz * dz);
        }
    }
}

// ---------------------------------------------------------------------------
// Kernel 2: gather sampled features, convert to f16 row-major (M x CIN) for
// WMMA A/B operands, and compute per-point squared norms (from the f16-rounded
// values so distances are self-consistent).
// ---------------------------------------------------------------------------
__global__ void gather_kernel(const float* __restrict__ feature,
                              const int* __restrict__ idx,
                              _Float16* __restrict__ Ah,
                              float* __restrict__ xx)
{
    const int g = blockIdx.x * blockDim.x + threadIdx.x;
    if (g >= B_ * M_) return;
    const int b = g / M_;
    const int j = idx[g];
    const float* F = feature + (size_t)b * CIN_ * N_ + j;
    _Float16* ar = Ah + (size_t)g * CIN_;
    float ss = 0.0f;
    for (int c = 0; c < CIN_; ++c) {
        const _Float16 h = (_Float16)F[(size_t)c * N_];
        const float vh = (float)h;
        ss += vh * vh;
        ar[c] = h;
    }
    xx[g] = ss;
}

// weightbank (64 x 512) -> f16 transposed (512 x 64) so WMMA B-operand loads
// are contiguous over K.
__global__ void wbconv_kernel(const float* __restrict__ wb, _Float16* __restrict__ WbT)
{
    const int e = blockIdx.x * blockDim.x + threadIdx.x;
    if (e >= 512 * CIN_) return;
    const int o = e >> 6, c = e & 63;
    WbT[o * CIN_ + c] = (_Float16)wb[c * 512 + o];
}

// ---------------------------------------------------------------------------
// WMMA fragment load helpers (contiguous k order for A and B, wave32).
// A (16-bit 16x32, MxK): lane m = lane&15, g = lane>>4;
//   halves 0..7  hold K = k0 + 16g .. k0+16g+7      (contiguous)
//   halves 8..15 hold K = k0+16g+8 .. k0+16g+15  (contiguous)
// B (16-bit 32x16, KxN): lane n = lane&15, g = lane>>4;
//   halves 0..15 hold K = k0 + 16g .. k0+16g+15   (contiguous)
// ---------------------------------------------------------------------------
__device__ __forceinline__ v16h load_a_frag(const _Float16* rowptr, int k0, int gsel)
{
    const v8h lo = *(const v8h*)(rowptr + k0 + 16 * gsel);
    const v8h hi = *(const v8h*)(rowptr + k0 + 16 * gsel + 8);
    v16h a;
#pragma unroll
    for (int q = 0; q < 8; ++q) { a[q] = lo[q]; a[q + 8] = hi[q]; }
    return a;
}

// ---------------------------------------------------------------------------
// Kernel 3: kNN pairwise scores. pd[i,j] = 2*(f_i . f_j) - |f_i|^2 - |f_j|^2.
// 4 waves/block, one 16x16 output tile per wave, K=64 as two x32 WMMA steps.
// ---------------------------------------------------------------------------
__global__ __launch_bounds__(128) void knn_gemm_kernel(const _Float16* __restrict__ Ah,
                                                       const float* __restrict__ xx,
                                                       float* __restrict__ pd)
{
    const int b = blockIdx.z;
    const int i0 = blockIdx.x * 16;
    const int wave = threadIdx.x >> 5, lane = threadIdx.x & 31;
    const int j0 = (blockIdx.y * 4 + wave) * 16;
    const int n = lane & 15, gsel = lane >> 4;
    const _Float16* Ab = Ah + (size_t)b * M_ * CIN_;

    v8f acc = {};
#pragma unroll
    for (int k0 = 0; k0 < CIN_; k0 += 32) {
        const v16h a  = load_a_frag(Ab + (size_t)(i0 + n) * CIN_, k0, gsel);
        const v16h bm = *(const v16h*)(Ab + (size_t)(j0 + n) * CIN_ + k0 + 16 * gsel);
        acc = __builtin_amdgcn_wmma_f32_16x16x32_f16(false, a, false, bm,
                                                     (short)0, acc, false, false);
    }
    const float xj = xx[b * M_ + j0 + n];
#pragma unroll
    for (int r = 0; r < 8; ++r) {
        const int row = i0 + r + 8 * gsel;   // C/D layout: VGPR r -> M = r + 8*(lane/16)
        pd[((size_t)b * M_ + row) * M_ + j0 + n] =
            2.0f * acc[r] - xx[b * M_ + row] - xj;
    }
}

// ---------------------------------------------------------------------------
// Kernel 4: top-16 per row (register-resident insertion list, fully unrolled).
// ---------------------------------------------------------------------------
__global__ void topk_kernel(const float* __restrict__ pd, int* __restrict__ kidx)
{
    const int g = blockIdx.x * blockDim.x + threadIdx.x;
    if (g >= B_ * M_) return;
    const float* row = pd + (size_t)g * M_;
    float bv[KNN_]; int bj[KNN_];
#pragma unroll
    for (int q = 0; q < KNN_; ++q) { bv[q] = -INFINITY; bj[q] = 0; }
    for (int j = 0; j < M_; ++j) {
        const float v = row[j];
        if (v > bv[KNN_ - 1]) {
            bv[KNN_ - 1] = v; bj[KNN_ - 1] = j;
#pragma unroll
            for (int q = KNN_ - 1; q > 0; --q) {
                if (bv[q] > bv[q - 1]) {
                    const float tv = bv[q]; bv[q] = bv[q - 1]; bv[q - 1] = tv;
                    const int   tj = bj[q]; bj[q] = bj[q - 1]; bj[q - 1] = tj;
                }
            }
        }
    }
#pragma unroll
    for (int q = 0; q < KNN_; ++q) kidx[(size_t)g * KNN_ + q] = bj[q];
}

// ---------------------------------------------------------------------------
// Kernel 5: scorenet MLP (64 -> relu 16 -> 8 -> softmax) per (point, neighbor).
// Note reference quirk: gathers from the ORIGINAL feature tensor's first M
// columns (k_idx values are in [0, M)).
// ---------------------------------------------------------------------------
__global__ void scorenet_kernel(const float* __restrict__ feature,
                                const int* __restrict__ kidx,
                                const float* __restrict__ w1, const float* __restrict__ b1,
                                const float* __restrict__ w2, const float* __restrict__ b2,
                                float* __restrict__ scores)
{
    const int g = blockIdx.x * blockDim.x + threadIdx.x;
    if (g >= B_ * M_ * KNN_) return;
    const int b = g / (M_ * KNN_);
    const int j = kidx[g];

    float gv[CIN_];
    const float* F = feature + (size_t)b * CIN_ * N_ + j;
    for (int c = 0; c < CIN_; ++c) gv[c] = F[(size_t)c * N_];

    float logit[MKER_];
#pragma unroll
    for (int mm = 0; mm < MKER_; ++mm) logit[mm] = b2[mm];
    for (int hh = 0; hh < SNH_; ++hh) {
        float a = b1[hh];
        for (int c = 0; c < CIN_; ++c) a += gv[c] * w1[c * SNH_ + hh];
        a = fmaxf(a, 0.0f);
#pragma unroll
        for (int mm = 0; mm < MKER_; ++mm) logit[mm] += a * w2[hh * MKER_ + mm];
    }
    float mx = logit[0];
#pragma unroll
    for (int mm = 1; mm < MKER_; ++mm) mx = fmaxf(mx, logit[mm]);
    float sum = 0.0f;
#pragma unroll
    for (int mm = 0; mm < MKER_; ++mm) { logit[mm] = __expf(logit[mm] - mx); sum += logit[mm]; }
    const float inv = 1.0f / sum;
#pragma unroll
    for (int mm = 0; mm < MKER_; ++mm) scores[(size_t)g * MKER_ + mm] = logit[mm] * inv;
}

// ---------------------------------------------------------------------------
// Kernel 6: feat_trans GEMM: xt(B,M,512) = featT(M x 64) @ weightbank(64 x 512).
// Same WMMA tiling; B operand from the transposed f16 weightbank.
// ---------------------------------------------------------------------------
__global__ __launch_bounds__(128) void trans_gemm_kernel(const _Float16* __restrict__ Ah,
                                                         const _Float16* __restrict__ WbT,
                                                         float* __restrict__ xt)
{
    const int b = blockIdx.z;
    const int i0 = blockIdx.x * 16;
    const int wave = threadIdx.x >> 5, lane = threadIdx.x & 31;
    const int o0 = (blockIdx.y * 4 + wave) * 16;
    const int n = lane & 15, gsel = lane >> 4;
    const _Float16* Ab = Ah + (size_t)b * M_ * CIN_;

    v8f acc = {};
#pragma unroll
    for (int k0 = 0; k0 < CIN_; k0 += 32) {
        const v16h a  = load_a_frag(Ab + (size_t)(i0 + n) * CIN_, k0, gsel);
        const v16h bm = *(const v16h*)(WbT + (size_t)(o0 + n) * CIN_ + k0 + 16 * gsel);
        acc = __builtin_amdgcn_wmma_f32_16x16x32_f16(false, a, false, bm,
                                                     (short)0, acc, false, false);
    }
#pragma unroll
    for (int r = 0; r < 8; ++r) {
        const int row = i0 + r + 8 * gsel;
        xt[((size_t)b * M_ + row) * (MKER_ * COUT_) + o0 + n] = acc[r];
    }
}

// ---------------------------------------------------------------------------
// Kernel 7: assemble feats[b,c,i] = sum_{k,m} scores[b,i,k,m] * xt[b,kidx,m,c]
// and accumulate per-channel batch stats with atomics.
// One 64-thread block per (i,b); thread = output channel; xt rows coalesced.
// ---------------------------------------------------------------------------
__global__ __launch_bounds__(64) void assemble_kernel(const float* __restrict__ xt,
                                                      const float* __restrict__ scores,
                                                      const int* __restrict__ kidx,
                                                      float* __restrict__ feats,
                                                      float* __restrict__ sums)
{
    const int b = blockIdx.y, i = blockIdx.x, c = threadIdx.x;
    __shared__ float sc[KNN_ * MKER_];
    __shared__ int   ki[KNN_];
    const size_t rbase = (size_t)b * M_ + i;
    for (int q = c; q < KNN_ * MKER_; q += 64) sc[q] = scores[rbase * KNN_ * MKER_ + q];
    if (c < KNN_) ki[c] = kidx[rbase * KNN_ + c];
    __syncthreads();

    float acc = 0.0f;
    for (int k = 0; k < KNN_; ++k) {
        const float* xr = xt + ((size_t)b * M_ + ki[k]) * (MKER_ * COUT_) + c;
#pragma unroll
        for (int m = 0; m < MKER_; ++m)
            acc += sc[k * MKER_ + m] * xr[m * COUT_];
    }
    feats[((size_t)b * COUT_ + c) * M_ + i] = acc;
    atomicAdd(&sums[c], acc);
    atomicAdd(&sums[COUT_ + c], acc * acc);
}

// BN stats -> per-channel scale/shift
__global__ void bn_stats_kernel(const float* __restrict__ sums,
                                const float* __restrict__ gamma,
                                const float* __restrict__ beta,
                                float* __restrict__ bnp)
{
    const int c = threadIdx.x;
    const float cnt = (float)(B_ * M_);
    const float mu = sums[c] / cnt;
    const float var = sums[COUT_ + c] / cnt - mu * mu;
    const float sc = gamma[c] * rsqrtf(var + 1e-5f);
    bnp[c] = sc;
    bnp[COUT_ + c] = beta[c] - mu * sc;
}

// normalize + ReLU into the output
__global__ void bn_apply_kernel(const float* __restrict__ feats,
                                const float* __restrict__ bnp,
                                float* __restrict__ out)
{
    const int e = blockIdx.x * blockDim.x + threadIdx.x;
    if (e >= B_ * COUT_ * M_) return;
    const int c = (e / M_) % COUT_;
    const float v = feats[e] * bnp[c] + bnp[COUT_ + c];
    out[e] = v > 0.0f ? v : 0.0f;
}

// ---------------------------------------------------------------------------
extern "C" void kernel_launch(void* const* d_in, const int* in_sizes, int n_in,
                              void* d_out, int out_size, void* d_ws, size_t ws_size,
                              hipStream_t stream)
{
    (void)in_sizes; (void)n_in; (void)out_size; (void)ws_size;
    const float* xyz     = (const float*)d_in[0];
    const float* feature = (const float*)d_in[1];
    const float* wb      = (const float*)d_in[2];
    const float* w1      = (const float*)d_in[3];
    const float* b1      = (const float*)d_in[4];
    const float* w2      = (const float*)d_in[5];
    const float* b2      = (const float*)d_in[6];
    const float* gamma   = (const float*)d_in[7];
    const float* beta    = (const float*)d_in[8];

    float* out       = (float*)d_out;
    float* p2        = out;                       // (B,M,3) = 24576 floats
    float* out_feats = out + B_ * M_ * 3;         // (B,COUT,M) = 524288 floats

    char* w = (char*)d_ws;
    size_t off = 0;
    auto alloc = [&](size_t bytes) -> void* {
        void* p = w + off;
        off = (off + bytes + 255) & ~(size_t)255;
        return p;
    };
    int*      idx    = (int*)     alloc((size_t)B_ * M_ * 4);
    _Float16* Ah     = (_Float16*)alloc((size_t)B_ * M_ * CIN_ * 2);
    float*    xx     = (float*)   alloc((size_t)B_ * M_ * 4);
    _Float16* WbT    = (_Float16*)alloc((size_t)512 * CIN_ * 2);
    float*    pd     = (float*)   alloc((size_t)B_ * M_ * M_ * 4);      // 67 MB
    int*      kidx   = (int*)     alloc((size_t)B_ * M_ * KNN_ * 4);
    float*    scores = (float*)   alloc((size_t)B_ * M_ * KNN_ * MKER_ * 4);
    float*    xt     = (float*)   alloc((size_t)B_ * M_ * MKER_ * COUT_ * 4);
    float*    feats  = (float*)   alloc((size_t)B_ * COUT_ * M_ * 4);
    float*    sums   = (float*)   alloc((size_t)2 * COUT_ * 4);
    float*    bnp    = (float*)   alloc((size_t)2 * COUT_ * 4);

    hipMemsetAsync(sums, 0, 2 * COUT_ * sizeof(float), stream);

    fps_kernel<<<B_, 1024, 0, stream>>>(xyz, idx, p2);
    gather_kernel<<<(B_ * M_ + 255) / 256, 256, 0, stream>>>(feature, idx, Ah, xx);
    wbconv_kernel<<<(512 * CIN_ + 255) / 256, 256, 0, stream>>>(wb, WbT);
    knn_gemm_kernel<<<dim3(M_ / 16, M_ / 64, B_), 128, 0, stream>>>(Ah, xx, pd);
    topk_kernel<<<(B_ * M_ + 255) / 256, 256, 0, stream>>>(pd, kidx);
    scorenet_kernel<<<(B_ * M_ * KNN_ + 255) / 256, 256, 0, stream>>>(
        feature, kidx, w1, b1, w2, b2, scores);
    trans_gemm_kernel<<<dim3(M_ / 16, 512 / 64, B_), 128, 0, stream>>>(Ah, WbT, xt);
    assemble_kernel<<<dim3(M_, B_), 64, 0, stream>>>(xt, scores, kidx, feats, sums);
    bn_stats_kernel<<<1, COUT_, 0, stream>>>(sums, gamma, beta, bnp);
    bn_apply_kernel<<<(B_ * COUT_ * M_ + 255) / 256, 256, 0, stream>>>(feats, bnp, out_feats);
}
